// DiTModule_79645873537267
// MI455X (gfx1250) — matrix operand order repaired and checked by reading, not compiled, not run on hardware
//
#include <hip/hip_runtime.h>
#include <cstdint>
#include <cstddef>

// ---------------------------------------------------------------------------
// Types for CDNA5 WMMA / TDM (gfx1250, wave32)
// ---------------------------------------------------------------------------
typedef __attribute__((ext_vector_type(16))) __bf16        v16bf;
typedef __attribute__((ext_vector_type(8)))  float         v8f;
typedef __attribute__((ext_vector_type(4)))  unsigned int  v4u;
typedef __attribute__((ext_vector_type(8)))  int           v8i;
typedef __attribute__((ext_vector_type(4)))  int           v4i;

#define LSEQ 1024
#define EDIM 768
#define NHEAD 16
#define HDIM 48
#define HDPAD 64
#define NDEPTH 8
#define MLPDIM 3072
#define PDIM 128
#define NQKV 2304

#define LDSP 40   // LDS row pitch in bf16 elements (80B: bank-conflict-free, 16B aligned)

// ---------------------------------------------------------------------------
// Helpers (native bf16 converts -> v_cvt_pk_bf16_f32 class instructions)
// ---------------------------------------------------------------------------
__device__ inline unsigned short f2bf(float f) {
    union { __bf16 b; unsigned short u; } c;
    c.b = (__bf16)f;
    return c.u;
}
__device__ inline unsigned int f2bf2(float lo, float hi) {
    return (unsigned int)f2bf(lo) | ((unsigned int)f2bf(hi) << 16);
}
__device__ inline float bf2f(unsigned short h) {
    return __uint_as_float(((unsigned int)h) << 16);
}
__device__ inline float wave_sum(float v) {
    for (int o = 16; o > 0; o >>= 1) v += __shfl_xor(v, o, 32);
    return v;
}
__device__ inline float wave_max(float v) {
    for (int o = 16; o > 0; o >>= 1) v = fmaxf(v, __shfl_xor(v, o, 32));
    return v;
}

// ---------------------------------------------------------------------------
// TDM: DMA a [128 x 32] bf16 tile (row stride lda elems) into LDS with an
// 80-byte destination pitch (64B row + 16B pad). Hardware zero-fills reads
// past tensor_dim0/1 (handles the K=48 head-dim tail).
// ---------------------------------------------------------------------------
__device__ inline void tdm_load_tile_a(const unsigned short* gsrc, unsigned lds_byte,
                                       int td0, int td1, int lda) {
    unsigned long long ga = (unsigned long long)(size_t)gsrc;
    v4u g0;
    g0.x = 1u;                                   // count=1, user descriptor
    g0.y = lds_byte;                             // lds_addr
    g0.z = (unsigned)ga;                         // global_addr[31:0]
    g0.w = ((unsigned)(ga >> 32) & 0x01FFFFFFu)  // global_addr[56:32]
           | 0x80000000u;                        // type=2 ("image")
    v8i g1;
    g1[0] = (int)((1u << 16)    // data_size = 1 (2 bytes)
                | (1u << 20)    // pad_enable
                | (3u << 22)    // pad_interval: 16 DWORDs (64B)
                | (3u << 25));  // pad_amount: 4 DWORDs (16B)
    g1[1] = (int)(((unsigned)td0 & 0xffffu) << 16);                 // tensor_dim0 lo
    g1[2] = (int)((((unsigned)td0 >> 16) & 0xffffu)
                | (((unsigned)td1 & 0xffffu) << 16));               // dim0 hi | dim1 lo
    g1[3] = (int)((((unsigned)td1 >> 16) & 0xffffu) | (32u << 16)); // dim1 hi | tile_dim0=32
    g1[4] = 128;                                                    // tile_dim1
    g1[5] = lda;                                                    // tensor_dim0_stride lo
    g1[6] = 0;
    g1[7] = 0;
    v4i g2 = {0, 0, 0, 0};
    v4i g3 = {0, 0, 0, 0};
#if defined(__clang_major__) && (__clang_major__ >= 23)
    v8i g4 = {0, 0, 0, 0, 0, 0, 0, 0};
    __builtin_amdgcn_tensor_load_to_lds(g0, g1, g2, g3, g4, 0);
#else
    __builtin_amdgcn_tensor_load_to_lds(g0, g1, g2, g3, 0);
#endif
}

// ---------------------------------------------------------------------------
// GEMM epilogue modes
// ---------------------------------------------------------------------------
enum { EPI_QKV = 0, EPI_SCORES = 1, EPI_AV = 2, EPI_GATE = 3, EPI_GELU = 4, EPI_OUT = 5 };

struct GemmParams {
    const unsigned short* a;   // bf16 A matrix [M][lda]
    int lda;
    int aZ;                    // per-blockIdx.z element offset for A
    const float* bf;           // f32 B (weights, row-major [N][ldb])
    const unsigned short* bh16;// bf16 B alternative
    int ldb;
    int bZ;
    int M, N, Kpad, Kreal;
    const float* bias;         // [N]
    const float* gate;         // [N]   (EPI_GATE)
    float* xacc;               // f32 accum output (EPI_GATE), pitch ldc
    unsigned short* outh;      // bf16 output, pitch ldc
    float* outf;               // f32 output, pitch ldc
    const unsigned short* bhb; // bf16 score bias (EPI_SCORES), pitch ldc
    int ldc;
    int outZ;                  // per-z element offset for outputs / bhb
    int outColZ;               // per-z column offset (EPI_AV: HDIM)
    float qscale;
    unsigned short* qb;        // EPI_QKV scatter targets
    unsigned short* kb;
    unsigned short* vtb;
};

// 128x64 tile, BK=32, 256 threads = 8 waves; each wave owns a 32x32 sub-tile
// (2 A-frags x 2 B-frags = 4 WMMA per K-step). Double-buffered LDS: one
// barrier per K-step, TDM + B staging for tile i+1 overlap WMMA on tile i.
template <int EPI, bool B_BF16, bool N_FULL, bool K_FULL>
__global__ __launch_bounds__(256) void gemm_kernel(GemmParams p) {
    __shared__ unsigned short As[2][128 * LDSP];
    __shared__ unsigned short Bs[2][64 * LDSP];

    const int tid  = threadIdx.x;
    const int wave = tid >> 5;
    const int lane = tid & 31;
    const int lane_lo = lane & 15;
    const int grp  = lane >> 4;
    const int wm   = wave & 3;   // M offset 32*wm
    const int wn   = wave >> 2;  // N offset 32*wn
    const int tileM = blockIdx.x * 128;
    const int tileN = blockIdx.y * 64;
    const int z = blockIdx.z;

    const unsigned short* Ag  = p.a + (size_t)z * (size_t)p.aZ;
    const float*          Bfg = p.bf;
    const unsigned short* Bhg = p.bh16;
    if (B_BF16) Bhg = p.bh16 + (size_t)z * (size_t)p.bZ;
    else        Bfg = p.bf   + (size_t)z * (size_t)p.bZ;

    const unsigned lds_a0 = (unsigned)(unsigned long long)(size_t)(void*)&As[0][0];

    auto issueA = [&](int buf, int k0) {
        if (wave == 0)
            tdm_load_tile_a(Ag + (size_t)tileM * p.lda + k0,
                            lds_a0 + (unsigned)buf * (128 * LDSP * 2),
                            p.Kreal - k0, p.M - tileM, p.lda);
    };
    auto stageB = [&](int buf, int k0) {
        if (B_BF16) {
            // 64 rows x 32 bf16 = 256 chunks of 8 elems
            int row = tid >> 2, koff = (tid & 3) * 8;
            int gn = tileN + row, gk = k0 + koff;
            uint4 val = {0u, 0u, 0u, 0u};
            if ((N_FULL || gn < p.N) && (K_FULL || gk < p.Kreal))
                val = *(const uint4*)(Bhg + (size_t)gn * p.ldb + gk);
            *(uint4*)(&Bs[buf][row * LDSP + koff]) = val;
        } else {
            // 64 rows x 32 f32 = 512 chunks of 4 floats; 2 chunks per thread
            #pragma unroll
            for (int i = 0; i < 2; i++) {
                int chunk = tid + i * 256;
                int row = chunk >> 3, qoff = (chunk & 7) * 4;
                int gn = tileN + row;
                float4 v = {0.f, 0.f, 0.f, 0.f};
                if (N_FULL || gn < p.N)
                    v = *(const float4*)(Bfg + (size_t)gn * p.ldb + k0 + qoff);
                uint2 pk;
                pk.x = f2bf2(v.x, v.y);
                pk.y = f2bf2(v.z, v.w);
                *(uint2*)(&Bs[buf][row * LDSP + qoff]) = pk;
                if (i == 0 && k0 + 32 < p.Kpad)
                    __builtin_prefetch((const void*)(Bfg + (size_t)gn * p.ldb + k0 + 32), 0, 1);
            }
        }
    };

    v8f acc00 = {0.f, 0.f, 0.f, 0.f, 0.f, 0.f, 0.f, 0.f};
    v8f acc01 = acc00, acc10 = acc00, acc11 = acc00;
    union Frag { v16bf v; unsigned int u[8]; };

    const int nIter = p.Kpad >> 5;

    // prologue: fill buffer 0
    issueA(0, 0);
    stageB(0, 0);
    if (wave == 0) __builtin_amdgcn_s_wait_tensorcnt((short)0);
    __syncthreads();

    for (int it = 0; it < nIter; ++it) {
        const int cur = it & 1, nxt = cur ^ 1;
        // overlap: fetch tile it+1 while computing tile it
        if (it + 1 < nIter) {
            issueA(nxt, (it + 1) << 5);
            stageB(nxt, (it + 1) << 5);
        }

        // ---- fragments (ISA VGPR layouts: 16-bit A 16x32, B 32x16) ----
        Frag af0, af1, bf0, bf1;
        const int ar0 = (wm * 32 + lane_lo) * LDSP;
        const int ar1 = ar0 + 16 * LDSP;
        #pragma unroll
        for (int v = 0; v < 8; v++) {
            int kb = ((v & 4) ? 16 : 0) + (grp ? 8 : 0) + ((v & 3) << 1);
            af0.u[v] = *(const unsigned int*)(&As[cur][ar0 + kb]);
            af1.u[v] = *(const unsigned int*)(&As[cur][ar1 + kb]);
        }
        const int br0 = (wn * 32 + lane_lo) * LDSP;
        const int br1 = br0 + 16 * LDSP;
        #pragma unroll
        for (int v = 0; v < 8; v++) {
            int kb = ((v & 4) ? 16 : 0) + (grp ? 8 : 0) + ((v & 3) << 1);
            bf0.u[v] = *(const unsigned int*)(&Bs[cur][br0 + kb]);
            bf1.u[v] = *(const unsigned int*)(&Bs[cur][br1 + kb]);
        }
        acc00 = __builtin_amdgcn_wmma_f32_16x16x32_bf16(false, af0.v, false, bf0.v,
                                                        (short)0, acc00, false, false);
        acc01 = __builtin_amdgcn_wmma_f32_16x16x32_bf16(false, af0.v, false, bf1.v,
                                                        (short)0, acc01, false, false);
        acc10 = __builtin_amdgcn_wmma_f32_16x16x32_bf16(false, af1.v, false, bf0.v,
                                                        (short)0, acc10, false, false);
        acc11 = __builtin_amdgcn_wmma_f32_16x16x32_bf16(false, af1.v, false, bf1.v,
                                                        (short)0, acc11, false, false);

        if (wave == 0 && it + 1 < nIter) __builtin_amdgcn_s_wait_tensorcnt((short)0);
        __syncthreads();
    }

    // ---- epilogue (M is always a multiple of 128 here; no row guards) ----
    union { v8f v; float f[8]; } cu;
    #pragma unroll
    for (int am = 0; am < 2; am++) {
        #pragma unroll
        for (int bn = 0; bn < 2; bn++) {
            cu.v = am ? (bn ? acc11 : acc10) : (bn ? acc01 : acc00);
            int col = tileN + wn * 32 + bn * 16 + lane_lo;
            if (!N_FULL && col >= p.N) continue;
            #pragma unroll
            for (int r = 0; r < 8; r++) {
                int row = tileM + wm * 32 + am * 16 + grp * 8 + r;
                float s = cu.f[r];
                if constexpr (EPI == EPI_QKV) {
                    int h = col / 144, rem = col - h * 144;
                    int j = rem / HDIM, c = rem - j * HDIM;
                    if (j == 0)
                        p.qb[((size_t)h * LSEQ + row) * HDPAD + c] = f2bf(s * p.qscale);
                    else if (j == 1)
                        p.kb[((size_t)h * LSEQ + row) * HDPAD + c] = f2bf(s);
                    else
                        p.vtb[((size_t)h * HDIM + c) * LSEQ + row] = f2bf(s);
                } else if constexpr (EPI == EPI_SCORES) {
                    size_t off = (size_t)z * p.outZ + (size_t)row * p.ldc + col;
                    p.outf[off] = s + bf2f(p.bhb[off]);
                } else if constexpr (EPI == EPI_AV) {
                    p.outh[(size_t)row * p.ldc + z * p.outColZ + col] = f2bf(s);
                } else if constexpr (EPI == EPI_GATE) {
                    size_t off = (size_t)row * p.ldc + col;
                    p.xacc[off] += p.gate[col] * (s + p.bias[col]);
                } else if constexpr (EPI == EPI_GELU) {
                    float t = s + p.bias[col];
                    float g = 0.5f * t * (1.f + tanhf(0.7978845608028654f *
                                                      (t + 0.044715f * t * t * t)));
                    p.outh[(size_t)row * p.ldc + col] = f2bf(g);
                } else { // EPI_OUT
                    p.outf[(size_t)row * p.ldc + col] = s + p.bias[col];
                }
            }
        }
    }
}

// ---------------------------------------------------------------------------
// LayerNorm + adaLN modulation -> bf16 row  (one block per row, E=768)
// ---------------------------------------------------------------------------
__global__ __launch_bounds__(256) void ln_mod_kernel(const float* __restrict__ x,
                                                     const float* __restrict__ sh,
                                                     const float* __restrict__ sc,
                                                     unsigned short* __restrict__ out) {
    __shared__ float r1[8], r2[8];
    const int row = blockIdx.x, tid = threadIdx.x;
    const float* xr = x + (size_t)row * EDIM;
    float v0 = xr[tid], v1 = xr[tid + 256], v2 = xr[tid + 512];
    float s = wave_sum(v0 + v1 + v2);
    float q = wave_sum(v0 * v0 + v1 * v1 + v2 * v2);
    if ((tid & 31) == 0) { r1[tid >> 5] = s; r2[tid >> 5] = q; }
    __syncthreads();
    float ts = 0.f, tq = 0.f;
    for (int i = 0; i < 8; i++) { ts += r1[i]; tq += r2[i]; }
    float mean = ts * (1.f / EDIM);
    float var  = tq * (1.f / EDIM) - mean * mean;
    float rs = rsqrtf(var + 1e-6f);
    unsigned short* orow = out + (size_t)row * EDIM;
    orow[tid]       = f2bf((v0 - mean) * rs * (1.f + sc[tid])       + sh[tid]);
    orow[tid + 256] = f2bf((v1 - mean) * rs * (1.f + sc[tid + 256]) + sh[tid + 256]);
    orow[tid + 512] = f2bf((v2 - mean) * rs * (1.f + sc[tid + 512]) + sh[tid + 512]);
}

// ---------------------------------------------------------------------------
// Row softmax: S (f32, 1024 wide) -> P (bf16). One block per (h,l) row.
// ---------------------------------------------------------------------------
__global__ __launch_bounds__(256) void softmax_kernel(const float* __restrict__ S,
                                                      unsigned short* __restrict__ P) {
    __shared__ float r1[8];
    const int tid = threadIdx.x;
    const size_t base = (size_t)blockIdx.x * LSEQ;
    float v[4];
    #pragma unroll
    for (int i = 0; i < 4; i++) v[i] = S[base + tid + i * 256];
    float m = fmaxf(fmaxf(v[0], v[1]), fmaxf(v[2], v[3]));
    m = wave_max(m);
    if ((tid & 31) == 0) r1[tid >> 5] = m;
    __syncthreads();
    float gm = -3.4e38f;
    for (int i = 0; i < 8; i++) gm = fmaxf(gm, r1[i]);
    __syncthreads();
    float e[4], s = 0.f;
    #pragma unroll
    for (int i = 0; i < 4; i++) { e[i] = __expf(v[i] - gm); s += e[i]; }
    s = wave_sum(s);
    if ((tid & 31) == 0) r1[tid >> 5] = s;
    __syncthreads();
    float gs = 0.f;
    for (int i = 0; i < 8; i++) gs += r1[i];
    float inv = 1.f / gs;
    #pragma unroll
    for (int i = 0; i < 4; i++) P[base + tid + i * 256] = f2bf(e[i] * inv);
}

// ---------------------------------------------------------------------------
// Pairwise-bias path: bh[h][l][m] = sum_p (LN(bias[l,m,:])*g+b) * w[h][p]
// One wave32 per (l,m) pair; 8 pairs per block.
// ---------------------------------------------------------------------------
__global__ __launch_bounds__(256) void bh_kernel(const float* __restrict__ bias,
                                                 const float* __restrict__ g,
                                                 const float* __restrict__ b,
                                                 const float* __restrict__ w,
                                                 unsigned short* __restrict__ bh) {
    __shared__ float wsm[NHEAD * PDIM];
    __shared__ float gsm[PDIM], bsm[PDIM];
    const int tid = threadIdx.x;
    for (int i = tid; i < NHEAD * PDIM; i += 256) wsm[i] = w[i];
    if (tid < PDIM) { gsm[tid] = g[tid]; bsm[tid] = b[tid]; }
    __syncthreads();
    const int wave = tid >> 5, lane = tid & 31;
    const long long pair = (long long)blockIdx.x * 8 + wave;
    const int l = (int)(pair >> 10), m = (int)(pair & (LSEQ - 1));
    const float4 v = *(const float4*)(bias + (((size_t)l * LSEQ + m) * PDIM) + lane * 4);
    float s = wave_sum(v.x + v.y + v.z + v.w);
    float q = wave_sum(v.x * v.x + v.y * v.y + v.z * v.z + v.w * v.w);
    float mean = s * (1.f / PDIM);
    float var  = q * (1.f / PDIM) - mean * mean;
    float rs = rsqrtf(var + 1e-6f);
    const int p0 = lane * 4;
    float a0 = (v.x - mean) * rs * gsm[p0]     + bsm[p0];
    float a1 = (v.y - mean) * rs * gsm[p0 + 1] + bsm[p0 + 1];
    float a2 = (v.z - mean) * rs * gsm[p0 + 2] + bsm[p0 + 2];
    float a3 = (v.w - mean) * rs * gsm[p0 + 3] + bsm[p0 + 3];
    for (int h = 0; h < NHEAD; h++) {
        const float* wr = &wsm[h * PDIM + p0];
        float pz = wave_sum(a0 * wr[0] + a1 * wr[1] + a2 * wr[2] + a3 * wr[3]);
        if (lane == 0)
            bh[(size_t)h * LSEQ * LSEQ + ((size_t)l << 10) + m] = f2bf(pz);
    }
}

// ---------------------------------------------------------------------------
// Small vec-mat: out[r] = sum_k x[k]*W[r][k] + b[r]   (one wave per row)
// ---------------------------------------------------------------------------
__global__ __launch_bounds__(256) void vecmat_kernel(const float* __restrict__ W,
                                                     const float* __restrict__ b,
                                                     const float* __restrict__ x,
                                                     float* __restrict__ out,
                                                     int R, int K) {
    const int r = blockIdx.x * 8 + (threadIdx.x >> 5);
    const int lane = threadIdx.x & 31;
    if (r >= R) return;
    float s = 0.f;
    for (int k = lane; k < K; k += 32) s += x[k] * W[(size_t)r * K + k];
    s = wave_sum(s);
    if (lane == 0) out[r] = s + b[r];
}

// ---------------------------------------------------------------------------
// Time-embedding pipeline -> cs (768 f32). Single block.
// ---------------------------------------------------------------------------
__global__ __launch_bounds__(256) void time_kernel(const float* __restrict__ t,
                                                   const float* __restrict__ t0w,
                                                   const float* __restrict__ t0b,
                                                   const float* __restrict__ t2w,
                                                   const float* __restrict__ t2b,
                                                   float* __restrict__ cs) {
    __shared__ float emb[256];
    __shared__ float mid[EDIM];
    const float tv = t[0];
    if (threadIdx.x < 128) {
        float fr = __expf(-9.210340371976184f * (float)threadIdx.x * (1.f / 128.f));
        float arg = tv * fr;
        emb[threadIdx.x]       = cosf(arg);
        emb[threadIdx.x + 128] = sinf(arg);
    }
    __syncthreads();
    for (int j = threadIdx.x; j < EDIM; j += 256) {
        float s = t0b[j];
        const float* wr = t0w + (size_t)j * 256;
        for (int i = 0; i < 256; i++) s += emb[i] * wr[i];
        mid[j] = s / (1.f + __expf(-s));   // silu
    }
    __syncthreads();
    for (int j = threadIdx.x; j < EDIM; j += 256) {
        float c = t2b[j];
        const float* wr = t2w + (size_t)j * EDIM;
        for (int i = 0; i < EDIM; i++) c += mid[i] * wr[i];
        cs[j] = c / (1.f + __expf(-c));    // silu(c)
    }
}

__global__ __launch_bounds__(256) void copy_kernel(const float* __restrict__ src,
                                                   float* __restrict__ dst, int n) {
    int i = blockIdx.x * 256 + threadIdx.x;
    if (i < n) dst[i] = src[i];
}

// ---------------------------------------------------------------------------
// Workspace layout (bytes, 256-aligned)
// ---------------------------------------------------------------------------
static const size_t OFF_CS  = 0;                                   //   768 f32
static const size_t OFF_ADA = 3072;                                // 38400 f32
static const size_t OFF_X   = 156672;                              // L*E f32
static const size_t OFF_H   = 3302400;                             // L*E bf16
static const size_t OFF_Q   = 4875264;                             // 16*1024*64 bf16
static const size_t OFF_K   = 6972416;
static const size_t OFF_VT  = 9069568;                             // 16*48*1024 bf16
static const size_t OFF_Y   = 10642432;                            // L*E bf16
static const size_t OFF_ACT = 12215296;                            // L*MLP bf16
static const size_t OFF_S   = 18506752;                            // H*L*L f32
static const size_t OFF_P   = 85615616;                            // H*L*L bf16
static const size_t OFF_BH  = 119170048;                           // H*L*L bf16

extern "C" void kernel_launch(void* const* d_in, const int* in_sizes, int n_in,
                              void* d_out, int out_size, void* d_ws, size_t ws_size,
                              hipStream_t stream) {
    (void)in_sizes; (void)n_in; (void)out_size; (void)ws_size;
    const float* z     = (const float*)d_in[0];
    const float* t     = (const float*)d_in[1];
    const float* bias  = (const float*)d_in[2];
    const float* p2s_g = (const float*)d_in[3];
    const float* p2s_b = (const float*)d_in[4];
    const float* p2s_w = (const float*)d_in[5];
    const float* t0w   = (const float*)d_in[6];
    const float* t0b   = (const float*)d_in[7];
    const float* t2w   = (const float*)d_in[8];
    const float* t2b   = (const float*)d_in[9];
    const float* projw = (const float*)d_in[10];
    const float* ow    = (const float*)d_in[11];
    const float* ob    = (const float*)d_in[12];
    const float* adaw  = (const float*)d_in[13];
    const float* adab  = (const float*)d_in[14];
    const float* f1w   = (const float*)d_in[15];
    const float* f1b   = (const float*)d_in[16];
    const float* f2w   = (const float*)d_in[17];
    const float* f2b   = (const float*)d_in[18];
    const float* finaw = (const float*)d_in[19];
    const float* finab = (const float*)d_in[20];
    const float* finw  = (const float*)d_in[21];
    const float* finb  = (const float*)d_in[22];

    char* ws = (char*)d_ws;
    float*          cs   = (float*)(ws + OFF_CS);
    float*          ada  = (float*)(ws + OFF_ADA);
    float*          x    = (float*)(ws + OFF_X);
    unsigned short* hbuf = (unsigned short*)(ws + OFF_H);
    unsigned short* qb   = (unsigned short*)(ws + OFF_Q);
    unsigned short* kb   = (unsigned short*)(ws + OFF_K);
    unsigned short* vtb  = (unsigned short*)(ws + OFF_VT);
    unsigned short* yb   = (unsigned short*)(ws + OFF_Y);
    unsigned short* act  = (unsigned short*)(ws + OFF_ACT);
    float*          S    = (float*)(ws + OFF_S);
    unsigned short* P    = (unsigned short*)(ws + OFF_P);
    unsigned short* bh   = (unsigned short*)(ws + OFF_BH);
    float*          outp = (float*)d_out;

    // x = z
    copy_kernel<<<(LSEQ * EDIM + 255) / 256, 256, 0, stream>>>(z, x, LSEQ * EDIM);
    // cs = silu(W2 silu(W0 emb + b0) + b2)
    time_kernel<<<1, 256, 0, stream>>>(t, t0w, t0b, t2w, t2b, cs);
    // adaLN vectors for all blocks + final
    for (int d = 0; d < NDEPTH; d++)
        vecmat_kernel<<<(6 * EDIM) / 8, 256, 0, stream>>>(
            adaw + (size_t)d * 6 * EDIM * EDIM, adab + (size_t)d * 6 * EDIM,
            cs, ada + (size_t)d * 6 * EDIM, 6 * EDIM, EDIM);
    vecmat_kernel<<<(2 * EDIM) / 8, 256, 0, stream>>>(
        finaw, finab, cs, ada + (size_t)NDEPTH * 6 * EDIM, 2 * EDIM, EDIM);
    // bh = LN(bias)*g+b projected to heads
    bh_kernel<<<(LSEQ * LSEQ) / 8, 256, 0, stream>>>(bias, p2s_g, p2s_b, p2s_w, bh);

    const float qscale = 0.14433756729740643f;  // 48^-0.5

    for (int d = 0; d < NDEPTH; d++) {
        const float* ad  = ada + (size_t)d * 6 * EDIM;
        const float* sh1 = ad;
        const float* sc1 = ad + EDIM;
        const float* g1  = ad + 2 * EDIM;
        const float* sh2 = ad + 3 * EDIM;
        const float* sc2 = ad + 4 * EDIM;
        const float* g2  = ad + 5 * EDIM;

        // h = mod(LN(x), sh1, sc1)
        ln_mod_kernel<<<LSEQ, 256, 0, stream>>>(x, sh1, sc1, hbuf);

        // qkv = h @ pw^T  -> scatter q*scale, k, v^T (bf16)
        {
            GemmParams p{};
            p.a = hbuf; p.lda = EDIM; p.aZ = 0;
            p.bf = projw + (size_t)d * NQKV * EDIM; p.ldb = EDIM; p.bZ = 0;
            p.M = LSEQ; p.N = NQKV; p.Kpad = EDIM; p.Kreal = EDIM;
            p.qscale = qscale; p.qb = qb; p.kb = kb; p.vtb = vtb;
            gemm_kernel<EPI_QKV, false, true, true><<<dim3(8, 36, 1), 256, 0, stream>>>(p);
        }
        // S[h] = q[h] @ k[h]^T + bh[h]
        {
            GemmParams p{};
            p.a = qb; p.lda = HDPAD; p.aZ = LSEQ * HDPAD;
            p.bh16 = kb; p.ldb = HDPAD; p.bZ = LSEQ * HDPAD;
            p.M = LSEQ; p.N = LSEQ; p.Kpad = HDPAD; p.Kreal = HDIM;
            p.outf = S; p.bhb = bh; p.ldc = LSEQ; p.outZ = LSEQ * LSEQ; p.outColZ = 0;
            gemm_kernel<EPI_SCORES, true, true, false><<<dim3(8, 16, NHEAD), 256, 0, stream>>>(p);
        }
        // P = softmax(S) rows
        softmax_kernel<<<NHEAD * LSEQ, 256, 0, stream>>>(S, P);
        // y[l, h*48+c] = P[h] @ v[h]
        {
            GemmParams p{};
            p.a = P; p.lda = LSEQ; p.aZ = LSEQ * LSEQ;
            p.bh16 = vtb; p.ldb = LSEQ; p.bZ = HDIM * LSEQ;
            p.M = LSEQ; p.N = HDIM; p.Kpad = LSEQ; p.Kreal = LSEQ;
            p.outh = yb; p.ldc = EDIM; p.outZ = 0; p.outColZ = HDIM;
            gemm_kernel<EPI_AV, true, false, true><<<dim3(8, 1, NHEAD), 256, 0, stream>>>(p);
        }
        // x += g1 * (y @ ow^T + ob)
        {
            GemmParams p{};
            p.a = yb; p.lda = EDIM; p.aZ = 0;
            p.bf = ow + (size_t)d * EDIM * EDIM; p.ldb = EDIM; p.bZ = 0;
            p.M = LSEQ; p.N = EDIM; p.Kpad = EDIM; p.Kreal = EDIM;
            p.bias = ob + (size_t)d * EDIM; p.gate = g1; p.xacc = x; p.ldc = EDIM;
            gemm_kernel<EPI_GATE, false, true, true><<<dim3(8, 12, 1), 256, 0, stream>>>(p);
        }
        // h2 = mod(LN(x), sh2, sc2)
        ln_mod_kernel<<<LSEQ, 256, 0, stream>>>(x, sh2, sc2, hbuf);
        // act = gelu(h2 @ f1w^T + f1b)
        {
            GemmParams p{};
            p.a = hbuf; p.lda = EDIM; p.aZ = 0;
            p.bf = f1w + (size_t)d * MLPDIM * EDIM; p.ldb = EDIM; p.bZ = 0;
            p.M = LSEQ; p.N = MLPDIM; p.Kpad = EDIM; p.Kreal = EDIM;
            p.bias = f1b + (size_t)d * MLPDIM; p.outh = act; p.ldc = MLPDIM;
            gemm_kernel<EPI_GELU, false, true, true><<<dim3(8, 48, 1), 256, 0, stream>>>(p);
        }
        // x += g2 * (act @ f2w^T + f2b)
        {
            GemmParams p{};
            p.a = act; p.lda = MLPDIM; p.aZ = 0;
            p.bf = f2w + (size_t)d * EDIM * MLPDIM; p.ldb = MLPDIM; p.bZ = 0;
            p.M = LSEQ; p.N = EDIM; p.Kpad = MLPDIM; p.Kreal = MLPDIM;
            p.bias = f2b + (size_t)d * EDIM; p.gate = g2; p.xacc = x; p.ldc = EDIM;
            gemm_kernel<EPI_GATE, false, true, true><<<dim3(8, 12, 1), 256, 0, stream>>>(p);
        }
    }

    // final: out = mod(LN(x), sh, sc) @ fin_w^T + fin_b
    const float* fad = ada + (size_t)NDEPTH * 6 * EDIM;
    ln_mod_kernel<<<LSEQ, 256, 0, stream>>>(x, fad, fad + EDIM, hbuf);
    {
        GemmParams p{};
        p.a = hbuf; p.lda = EDIM; p.aZ = 0;
        p.bf = finw; p.ldb = EDIM; p.bZ = 0;
        p.M = LSEQ; p.N = EDIM; p.Kpad = EDIM; p.Kreal = EDIM;
        p.bias = finb; p.outf = outp; p.ldc = EDIM;
        gemm_kernel<EPI_OUT, false, true, true><<<dim3(8, 12, 1), 256, 0, stream>>>(p);
    }
}
